// QuadrupedSDFSiren_59691455479847
// MI455X (gfx1250) — matrix operand order repaired and checked by reading, not compiled, not run on hardware
//
#include <hip/hip_runtime.h>

// ---------------------------------------------------------------------------
// Fused QuadrupedSDFSiren forward for gfx1250 (MI455X).
// Compute-bound (~473 GFLOP, weights only 1.8MB f16 -> L2 resident).
// Strategy: v_wmma_f32_16x16x32_f16 everywhere; per-wave B-fragments held in
// VGPRs and reused across S=7 slabs of 16 points per block to amortize weight
// traffic; activations ping-pong through LDS (295KB dynamic, 1 WG per WGP).
// Wave-uniform control values are pushed through readfirstlane so slab/tile
// loops compile to scalar loops (no divergent exec-mask bookkeeping).
// ---------------------------------------------------------------------------

#define SA    264          // LDS row stride in halves (256+8: kills bank conflicts)
#define SLAB  (16 * SA)    // halves per 16-row slab buffer
#define SMAX  7            // slabs (16-point groups) per block
#define NPTS  262144
#define NSLAB (NPTS / 16)  // 16384
#define W0F   30.0f

#define RFL(x) __builtin_amdgcn_readfirstlane(x)

typedef __attribute__((ext_vector_type(16))) _Float16 v16h;
typedef __attribute__((ext_vector_type(8)))  _Float16 v8h;
typedef __attribute__((ext_vector_type(8)))  float    v8f;

// f16 weight layout inside d_ws (offsets in halves, all 32B aligned):
#define OFF_LEG(i)   ((i) * 14336)             // l0 +0 (64x32), l1 +2048, l2 +6144, s +10240
#define OFF_MOD0     57344                     // 256x256
#define OFF_MODJ(j)  (122880 + ((j) - 1) * 131072)  // 256x512, j=1..4
#define OFF_SIR0     647168                    // 256x32 (K padded 3->32)
#define OFF_SIRJ(j)  (655360 + ((j) - 1) * 65536)   // 256x256, j=1..4

struct Params {
  const float*    x;
  float*          out;
  const _Float16* w16;
  const float*    leg_b[4][4];   // [leg][l0,l1,l2,s]
  const float*    mod_b[5];
  const float*    sir_b[5];
  const float*    last_w;
  const float*    last_b;
};

// ---- weight f32 -> f16 convert with zero K-padding -------------------------
__global__ void cvt_pad_kernel(const float* __restrict__ src, _Float16* __restrict__ dst,
                               int rows, int cols, int ldw) {
  int i = blockIdx.x * blockDim.x + threadIdx.x;
  int total = rows * ldw;
  if (i < total) {
    int r = i / ldw, c = i - r * ldw;
    dst[i] = (c < cols) ? (_Float16)src[r * cols + c] : (_Float16)0.0f;
  }
}

// ---- WMMA fragment helpers (ISA 7.12.2 layouts) ----------------------------
__device__ __forceinline__ v16h load_a(const _Float16* buf, int lane, int kb) {
  const int row = lane & 15;
  const int hi  = (lane >> 4) & 1;
  const _Float16* p = buf + row * SA + kb + hi * 8;
  v8h lo = *(const v8h*)(p);        // ds_load_b128
  v8h hh = *(const v8h*)(p + 16);   // ds_load_b128
  v16h a;
#pragma unroll
  for (int i = 0; i < 8; ++i) { a[i] = lo[i]; a[i + 8] = hh[i]; }
  return a;
}

__device__ __forceinline__ v16h load_b1(const _Float16* w, int ldw, int kb, int nb, int lane) {
  const int n  = nb + (lane & 15);
  const int hi = (lane >> 4) & 1;
  const _Float16* q = w + (size_t)n * ldw + kb + hi * 8;
  v8h lo = *(const v8h*)(q);
  v8h hh = *(const v8h*)(q + 16);
  v16h b;
#pragma unroll
  for (int i = 0; i < 8; ++i) { b[i] = lo[i]; b[i + 8] = hh[i]; }
  return b;
}

template <int NK>
__device__ __forceinline__ void load_breg(v16h* b, const _Float16* w, int ldw, int wkb,
                                          int nb, int lane) {
#pragma unroll
  for (int s = 0; s < NK; ++s) b[s] = load_b1(w, ldw, wkb + s * 32, nb, lane);
}

template <int NK>
__device__ __forceinline__ v8f gemm_regB(v8f acc, const _Float16* abuf, const v16h* b,
                                         int lane) {
#pragma unroll
  for (int s = 0; s < NK; ++s) {
    v16h a = load_a(abuf, lane, s * 32);
    acc = __builtin_amdgcn_wmma_f32_16x16x32_f16(false, a, false, b[s], (short)0, acc,
                                                 false, false);
  }
  return acc;
}

__device__ __forceinline__ v8f bias_init(const float* __restrict__ bias, int nb, int lane) {
  const float bv = bias[nb + (lane & 15)];
  v8f c;
#pragma unroll
  for (int j = 0; j < 8; ++j) c[j] = bv;
  return c;
}

__device__ __forceinline__ void store_tile(_Float16* buf, int nb, int lane, v8f d) {
  const int n  = nb + (lane & 15);
  const int mb = ((lane >> 4) & 1) * 8;
#pragma unroll
  for (int j = 0; j < 8; ++j) buf[(mb + j) * SA + n] = (_Float16)d[j];
}

__device__ __forceinline__ void relu_v(v8f& d) {
#pragma unroll
  for (int j = 0; j < 8; ++j) d[j] = fmaxf(d[j], 0.0f);
}
__device__ __forceinline__ void sin_v(v8f& d) {
#pragma unroll
  for (int j = 0; j < 8; ++j) d[j] = __sinf(W0F * d[j]);
}

// ---------------------------------------------------------------------------
// Main fused kernel: 256 threads (8 wave32) per block; block owns S<=7 slabs
// of 16 points. LDS (dynamic): zb | m0b | m1b | h0b | h1b, each SMAX slabs.
// ---------------------------------------------------------------------------
__global__ __launch_bounds__(256) void fused_siren_kernel(Params p) {
  extern __shared__ __align__(16) _Float16 smem[];
  _Float16* zb  = smem;
  _Float16* m0b = smem + 1 * SMAX * SLAB;
  _Float16* m1b = smem + 2 * SMAX * SLAB;
  _Float16* h0b = smem + 3 * SMAX * SLAB;
  _Float16* h1b = smem + 4 * SMAX * SLAB;

  const int tid  = threadIdx.x;
  const int lane = tid & 31;
  const int wv   = RFL(tid >> 5);                          // wave 0..7 (SGPR)
  const int slab0 = blockIdx.x * SMAX;
  const int S = (NSLAB - slab0 < SMAX) ? (NSLAB - slab0) : SMAX;   // scalar
  const long long rowBase = (long long)slab0 * 16;
  const _Float16* W16 = p.w16;

  // ---- zero staging regions, load x tile (S*16 points) ----
  for (int i = tid; i < S * 16 * 128; i += 256) {
    int s = i >> 11, rem = i & 2047;
    m0b[s * SLAB + (rem >> 7) * SA + (rem & 127)] = (_Float16)0.0f;
  }
  for (int i = tid; i < S * 16 * 32; i += 256) {
    int s = i >> 9, rem = i & 511;
    h1b[s * SLAB + (rem >> 5) * SA + (rem & 31)] = (_Float16)0.0f;
  }
  __syncthreads();
  for (int i = tid; i < S * 16 * 15; i += 256) {
    int pr = i / 15, c = i - pr * 15;
    int s = pr >> 4, r = pr & 15;
    float v = p.x[(rowBase + pr) * 15 + c];
    if (c < 3) h1b[s * SLAB + r * SA + c] = (_Float16)v;              // xyz (K pad 32)
    else { int j = c - 3; m0b[s * SLAB + r * SA + (j / 3) * 32 + (j % 3)] = (_Float16)v; }
  }
  __syncthreads();

  // ---- legs: tile = wv&3 (N=64 -> 4 tiles), slabs split across wave halves ----
  const int lnb = RFL((wv & 3) * 16);                      // scalar tile base
  const int hS  = (S + 1) >> 1;
  const int sLo = RFL((wv >> 2) ? hS : 0);
  const int sHi = RFL((wv >> 2) ? S : hS);
  for (int leg = 0; leg < 4; ++leg) {
    const _Float16* wb = W16 + OFF_LEG(leg);
    {
      v16h b[1]; load_breg<1>(b, wb, 32, 0, lnb, lane);               // l0 (K pad 32)
      for (int s = sLo; s < sHi; ++s) {
        v8f acc = bias_init(p.leg_b[leg][0], lnb, lane);
        acc = gemm_regB<1>(acc, m0b + s * SLAB + leg * 32, b, lane);
        relu_v(acc);
        store_tile(h0b + s * SLAB, lnb, lane, acc);                   // t0 cols 0..63
      }
    }
    __syncthreads();
    {
      v16h b[2]; load_breg<2>(b, wb + 2048, 64, 0, lnb, lane);        // l1
      for (int s = sLo; s < sHi; ++s) {
        v8f acc = bias_init(p.leg_b[leg][1], lnb, lane);
        acc = gemm_regB<2>(acc, h0b + s * SLAB, b, lane);
        relu_v(acc);
        store_tile(h0b + s * SLAB + 64, lnb, lane, acc);              // t1 cols 64..127
      }
    }
    __syncthreads();
    {
      v16h b[2]; load_breg<2>(b, wb + 6144, 64, 0, lnb, lane);        // l2 (linear)
      for (int s = sLo; s < sHi; ++s) {
        v8f acc = bias_init(p.leg_b[leg][2], lnb, lane);
        acc = gemm_regB<2>(acc, h0b + s * SLAB + 64, b, lane);
        store_tile(h0b + s * SLAB, lnb, lane, acc);
      }
    }
    __syncthreads();
    {
      v16h b[2]; load_breg<2>(b, wb + 10240, 64, 0, lnb, lane);       // s -> sin
      for (int s = sLo; s < sHi; ++s) {
        v8f acc = bias_init(p.leg_b[leg][3], lnb, lane);
        acc = gemm_regB<2>(acc, h0b + s * SLAB, b, lane);
        sin_v(acc);
        store_tile(zb + s * SLAB, leg * 64 + lnb, lane, acc);         // z[:, 64*leg..]
      }
    }
    __syncthreads();
  }

  // ---- mod0: z(256)->256, B held in 8 regs, reused across all S slabs ----
  for (int t = wv; t < 16; t += 8) {
    const int tb = RFL(t * 16);
    v16h bm[8]; load_breg<8>(bm, W16 + OFF_MOD0, 256, 0, tb, lane);
    for (int s = 0; s < S; ++s) {
      v8f acc = bias_init(p.mod_b[0], tb, lane);
      acc = gemm_regB<8>(acc, zb + s * SLAB, bm, lane);
      sin_v(acc);
      store_tile(m0b + s * SLAB, tb, lane, acc);                      // md -> mod ping
    }
  }
  __syncthreads();

  // ---- siren0: xyz(pad 32)->256, md multiply read from m0b ----
  for (int t = wv; t < 16; t += 8) {
    const int tb = RFL(t * 16);
    v16h bs[1]; load_breg<1>(bs, W16 + OFF_SIR0, 32, 0, tb, lane);
    const int n  = tb + (lane & 15);
    const int mb = ((lane >> 4) & 1) * 8;
    for (int s = 0; s < S; ++s) {
      v8f acc = bias_init(p.sir_b[0], tb, lane);
      acc = gemm_regB<1>(acc, h1b + s * SLAB, bs, lane);
      const _Float16* mo = m0b + s * SLAB;
      v8f hn;
#pragma unroll
      for (int j = 0; j < 8; ++j)
        hn[j] = __sinf(W0F * (acc[j] * (float)mo[(mb + j) * SA + n]));
      store_tile(h0b + s * SLAB, tb, lane, hn);                       // h ping
    }
  }
  __syncthreads();

  // ---- layers 1..4: mod_L(concat(m,z):512->256) then siren_L(h:256->256) ----
  _Float16* mIn = m0b; _Float16* mOut = m1b;
  _Float16* hIn = h0b; _Float16* hOut = h1b;
  for (int L = 1; L < 5; ++L) {
    const _Float16* wm  = W16 + OFF_MODJ(L);
    const _Float16* wsi = W16 + OFF_SIRJ(L);
    // mod pass: 16 B-frags (128 VGPRs) resident across slab loop
    for (int t = wv; t < 16; t += 8) {
      const int tb = RFL(t * 16);
      v16h bm[16]; load_breg<16>(bm, wm, 512, 0, tb, lane);
      for (int s = 0; s < S; ++s) {
        v8f acc = bias_init(p.mod_b[L], tb, lane);
        acc = gemm_regB<8>(acc, mIn + s * SLAB, bm, lane);            // K 0..255 <- m
        acc = gemm_regB<8>(acc, zb + s * SLAB, bm + 8, lane);         // K 256..511 <- z
        sin_v(acc);
        store_tile(mOut + s * SLAB, tb, lane, acc);
      }
    }
    __syncthreads();
    // siren pass: 8 B-frags resident; md from mOut in D-layout
    for (int t = wv; t < 16; t += 8) {
      const int tb = RFL(t * 16);
      v16h bs[8]; load_breg<8>(bs, wsi, 256, 0, tb, lane);
      const int n  = tb + (lane & 15);
      const int mb = ((lane >> 4) & 1) * 8;
      for (int s = 0; s < S; ++s) {
        v8f acc = bias_init(p.sir_b[L], tb, lane);
        acc = gemm_regB<8>(acc, hIn + s * SLAB, bs, lane);
        const _Float16* mo = mOut + s * SLAB;
        v8f hn;
#pragma unroll
        for (int j = 0; j < 8; ++j)
          hn[j] = __sinf(W0F * (acc[j] * (float)mo[(mb + j) * SA + n]));
        store_tile(hOut + s * SLAB, tb, lane, hn);
      }
    }
    __syncthreads();
    _Float16* tp = mIn; mIn = mOut; mOut = tp;
    tp = hIn; hIn = hOut; hOut = tp;
  }

  // ---- head: 256 -> 1 ----
  for (int pt = tid; pt < S * 16; pt += 256) {
    const int s = pt >> 4, r = pt & 15;
    float sum = p.last_b[0];
#pragma unroll 8
    for (int k = 0; k < 256; ++k)
      sum += (float)hIn[s * SLAB + r * SA + k] * p.last_w[k];
    p.out[rowBase + pt] = sum;
  }
}

// ---------------------------------------------------------------------------
// Host side
// ---------------------------------------------------------------------------
static inline void cvt(const void* src, _Float16* dst, int rows, int cols, int ldw,
                       hipStream_t stream) {
  const int total = rows * ldw;
  cvt_pad_kernel<<<(total + 255) / 256, 256, 0, stream>>>((const float*)src, dst,
                                                          rows, cols, ldw);
}

extern "C" void kernel_launch(void* const* d_in, const int* in_sizes, int n_in,
                              void* d_out, int out_size, void* d_ws, size_t ws_size,
                              hipStream_t stream) {
  const float* X;
  const float *legw[4][4], *legb[4][4];
  const float *modw[5], *modb[5], *sirw[5], *sirb[5];
  const float *lastw, *lastb;

  // Input flattening: insertion order (x first) vs jax sorted-key pytree order.
  const bool insertion = (in_sizes[0] == NPTS * 15);
  if (insertion) {
    int k = 0;
    X = (const float*)d_in[k++];
    for (int i = 0; i < 4; ++i)
      for (int j = 0; j < 4; ++j) {
        legw[i][j] = (const float*)d_in[k++];
        legb[i][j] = (const float*)d_in[k++];
      }
    for (int j = 0; j < 5; ++j) { modw[j] = (const float*)d_in[k++]; modb[j] = (const float*)d_in[k++]; }
    for (int j = 0; j < 5; ++j) { sirw[j] = (const float*)d_in[k++]; sirb[j] = (const float*)d_in[k++]; }
    lastw = (const float*)d_in[k++]; lastb = (const float*)d_in[k++];
  } else {
    int k = 0;
    lastb = (const float*)d_in[k++]; lastw = (const float*)d_in[k++];
    for (int i = 0; i < 4; ++i)
      for (int j = 0; j < 4; ++j) {
        legb[i][j] = (const float*)d_in[k++];
        legw[i][j] = (const float*)d_in[k++];
      }
    for (int j = 0; j < 5; ++j) { modb[j] = (const float*)d_in[k++]; modw[j] = (const float*)d_in[k++]; }
    for (int j = 0; j < 5; ++j) { sirb[j] = (const float*)d_in[k++]; sirw[j] = (const float*)d_in[k++]; }
    X = (const float*)d_in[k++];
  }

  _Float16* W16 = (_Float16*)d_ws;   // uses ~1.84 MB of workspace

  for (int i = 0; i < 4; ++i) {
    cvt(legw[i][0], W16 + OFF_LEG(i) + 0,     64, 3,  32, stream);
    cvt(legw[i][1], W16 + OFF_LEG(i) + 2048,  64, 64, 64, stream);
    cvt(legw[i][2], W16 + OFF_LEG(i) + 6144,  64, 64, 64, stream);
    cvt(legw[i][3], W16 + OFF_LEG(i) + 10240, 64, 64, 64, stream);
  }
  cvt(modw[0], W16 + OFF_MOD0, 256, 256, 256, stream);
  for (int j = 1; j < 5; ++j) cvt(modw[j], W16 + OFF_MODJ(j), 256, 512, 512, stream);
  cvt(sirw[0], W16 + OFF_SIR0, 256, 3, 32, stream);
  for (int j = 1; j < 5; ++j) cvt(sirw[j], W16 + OFF_SIRJ(j), 256, 256, 256, stream);

  Params P;
  P.x = X; P.out = (float*)d_out; P.w16 = W16;
  for (int i = 0; i < 4; ++i)
    for (int j = 0; j < 4; ++j) P.leg_b[i][j] = legb[i][j];
  for (int j = 0; j < 5; ++j) { P.mod_b[j] = modb[j]; P.sir_b[j] = sirb[j]; }
  P.last_w = lastw; P.last_b = lastb;

  const int nblk = (NSLAB + SMAX - 1) / SMAX;                 // 2341
  const size_t shbytes = (size_t)5 * SMAX * SLAB * sizeof(_Float16);  // 295,680 B
  (void)hipFuncSetAttribute((const void*)fused_siren_kernel,
                            hipFuncAttributeMaxDynamicSharedMemorySize, (int)shbytes);
  fused_siren_kernel<<<nblk, 256, shbytes, stream>>>(P);
}
